// GraphAttention_13262859010444
// MI455X (gfx1250) — compile-verified, matrix operand order checked by reading, not run on hardware
//
#include <hip/hip_runtime.h>
#include <hip/hip_bf16.h>

// GraphAttention on gfx1250 (MI455X, wave32, WMMA).
//
//   wx = x @ W            -> phase 1 (memory-bound, VALU FMA), stored f16 as wx2[b][e][j]
//   s, masked softmax     -> phase 2 (tiny)
//   out[b] = att @ wx[b]  -> phase 3: single GEMM M=1024,K=1024,N=65536 using
//                            v_wmma_f32_16x16x32_f16 (f32 accumulate)

typedef __attribute__((ext_vector_type(16))) _Float16 v16h;
typedef __attribute__((ext_vector_type(8)))  _Float16 v8h;
typedef __attribute__((ext_vector_type(8)))  float    v8f;

#define NN     1024
#define C2C    64
#define EEC    64
#define ALPHA  0.2f
#define NEGF   (-1e12f)

// ---------------------------------------------------------------------------
// Phase 1: wx[b,n,e] = sum_c x[b,n,c] * W[c,e]
//          s[b,n]    = sum_e lrelu(wx[b,n,e]) * a[e]
//          wx2[b][e][n] = (f16) wx[b,n,e]      (K-contiguous layout for GEMM B)
// ---------------------------------------------------------------------------
__global__ __launch_bounds__(256) void k_wx(const float* __restrict__ x,
                                            const float* __restrict__ W,
                                            const float* __restrict__ avec,
                                            _Float16* __restrict__ wx2,
                                            float* __restrict__ svec) {
  __shared__ float Ws[64][64];     // W row-major [c][e]
  __shared__ float xs[64][68];     // x tile [n][c], padded
  __shared__ float red[4][64];

  const int t  = threadIdx.x;
  const int eg = t >> 6;           // 0..3, uniform per wave32
  const int nl = t & 63;
  const int b  = blockIdx.y;
  const int n0 = blockIdx.x * 64;

#pragma unroll
  for (int q = 0; q < 16; ++q) {
    int idx = q * 256 + t;
    Ws[idx >> 6][idx & 63] = W[idx];
  }
  const float* xrow = x + ((size_t)b * NN + n0 + nl) * C2C;
#pragma unroll
  for (int q = 0; q < 4; ++q) {
    *(float4*)&xs[nl][eg * 16 + q * 4] = *(const float4*)&xrow[eg * 16 + q * 4];
  }
  __syncthreads();

  float acc[16];
#pragma unroll
  for (int i = 0; i < 16; ++i) acc[i] = 0.0f;

#pragma unroll 8
  for (int c = 0; c < 64; ++c) {
    float xv = xs[nl][c];
#pragma unroll
    for (int i = 0; i < 16; ++i)
      acc[i] = fmaf(xv, Ws[c][eg * 16 + i], acc[i]);  // Ws read is wave-uniform
  }

  float sacc = 0.0f;
#pragma unroll
  for (int i = 0; i < 16; ++i) {
    const int e = eg * 16 + i;
    const float v  = acc[i];
    const float lr = v > 0.0f ? v : ALPHA * v;
    sacc = fmaf(lr, avec[e], sacc);
    wx2[((size_t)b * EEC + e) * NN + n0 + nl] = (_Float16)v;  // coalesced in n
  }
  red[eg][nl] = sacc;
  __syncthreads();
  if (t < 64) {
    svec[(size_t)b * NN + n0 + t] = red[0][t] + red[1][t] + red[2][t] + red[3][t];
  }
}

// ---------------------------------------------------------------------------
// Phase 2: att[i,:] = softmax_j( s[i,j] + (adj[i,j] > 0 ? 0 : -1e12) ), f16 out
// ---------------------------------------------------------------------------
__global__ __launch_bounds__(256) void k_softmax(const float* __restrict__ svec,
                                                 const float* __restrict__ adj,
                                                 _Float16* __restrict__ att) {
  __shared__ float red[256];
  const int t = threadIdx.x;
  const int i = blockIdx.x;

  float v[4];
  float mx = -__builtin_inff();
#pragma unroll
  for (int q = 0; q < 4; ++q) {
    const size_t j = (size_t)i * NN + q * 256 + t;
    float sc = svec[j] + (adj[j] > 0.0f ? 0.0f : NEGF);
    v[q] = sc;
    mx = fmaxf(mx, sc);
  }
  red[t] = mx;
  __syncthreads();
  for (int s = 128; s > 0; s >>= 1) {
    if (t < s) red[t] = fmaxf(red[t], red[t + s]);
    __syncthreads();
  }
  mx = red[0];
  __syncthreads();

  float sum = 0.0f;
#pragma unroll
  for (int q = 0; q < 4; ++q) { v[q] = __expf(v[q] - mx); sum += v[q]; }
  red[t] = sum;
  __syncthreads();
  for (int s = 128; s > 0; s >>= 1) {
    if (t < s) red[t] += red[t + s];
    __syncthreads();
  }
  const float inv = 1.0f / red[0];
#pragma unroll
  for (int q = 0; q < 4; ++q)
    att[(size_t)i * NN + q * 256 + t] = (_Float16)(v[q] * inv);
}

// ---------------------------------------------------------------------------
// Phase 3: C[i, col] = sum_j att[i,j] * wx2[col][j],   col = b*64 + e
//          out[b][i][e] = C[i, col] + bias[e]
// Block tile 128x128, BK=32, 8 waves (2 in M x 4 in N), each wave: 4x2 WMMA
// tiles of 16x16 -> v_wmma_f32_16x16x32_f16.
// ---------------------------------------------------------------------------
__global__ __launch_bounds__(256) void k_gemm(const _Float16* __restrict__ att,
                                              const _Float16* __restrict__ wx2,
                                              const float* __restrict__ bias,
                                              float* __restrict__ out) {
  __shared__ __align__(16) _Float16 As[128][40];  // [m][k], +8 pad keeps 16B align
  __shared__ __align__(16) _Float16 Bs[128][40];  // [n][k]

  const int t    = threadIdx.x;
  const int lane = t & 31;
  const int wave = t >> 5;
  const int wm   = wave & 1;   // 64-row slab
  const int wn   = wave >> 1;  // 32-col slab
  const int m0   = blockIdx.y * 128;
  const int c0   = blockIdx.x * 128;
  const int l    = lane & 15;
  const int h    = lane >> 4;  // half-wave select per ISA 16-bit A layout

  v8f acc[4][2];
#pragma unroll
  for (int mt = 0; mt < 4; ++mt)
#pragma unroll
    for (int nt = 0; nt < 2; ++nt)
#pragma unroll
      for (int r = 0; r < 8; ++r) acc[mt][nt][r] = 0.0f;

  // staging mapping: 2 x uint4 (8 halves) per thread per matrix
  const int srow  = t >> 2;
  const int skoff = (t & 3) * 8;

  for (int k0 = 0; k0 < NN; k0 += 32) {
#pragma unroll
    for (int q = 0; q < 2; ++q) {
      const int row  = srow + q * 64;
      *(uint4*)&As[row][skoff] =
          *(const uint4*)&att[(size_t)(m0 + row) * NN + k0 + skoff];
      *(uint4*)&Bs[row][skoff] =
          *(const uint4*)&wx2[(size_t)(c0 + row) * NN + k0 + skoff];
    }
    if (k0 + 32 < NN) {  // hint next K-tile into cache (global_prefetch_b8)
      __builtin_prefetch(&att[(size_t)(m0 + srow) * NN + k0 + 32 + skoff], 0, 1);
      __builtin_prefetch(&wx2[(size_t)(c0 + srow) * NN + k0 + 32 + skoff], 0, 1);
    }
    __syncthreads();

    // Fragment gather per ISA: lane(h=lane>>4) holds K = h*8+0..7 (VGPR0-3)
    // and K = 16+h*8+0..7 (VGPR4-7); A lane row = lane&15, B lane col = lane&15.
    v16h af[4], bf[2];
#pragma unroll
    for (int mt = 0; mt < 4; ++mt) {
      const int m = wm * 64 + mt * 16 + l;
      v8h lo = *(const v8h*)&As[m][h * 8];
      v8h hi = *(const v8h*)&As[m][16 + h * 8];
#pragma unroll
      for (int i = 0; i < 8; ++i) { af[mt][i] = lo[i]; af[mt][i + 8] = hi[i]; }
    }
#pragma unroll
    for (int nt = 0; nt < 2; ++nt) {
      const int n = wn * 32 + nt * 16 + l;
      v8h lo = *(const v8h*)&Bs[n][h * 8];
      v8h hi = *(const v8h*)&Bs[n][16 + h * 8];
#pragma unroll
      for (int i = 0; i < 8; ++i) { bf[nt][i] = lo[i]; bf[nt][i + 8] = hi[i]; }
    }

#pragma unroll
    for (int mt = 0; mt < 4; ++mt)
#pragma unroll
      for (int nt = 0; nt < 2; ++nt)
        acc[mt][nt] = __builtin_amdgcn_wmma_f32_16x16x32_f16(
            false, af[mt], false, bf[nt], (short)0, acc[mt][nt], false, false);
    __syncthreads();
  }

  // Store: C/D layout -> lane l = N column, row m = h*8 + r; fuse bias add.
#pragma unroll
  for (int nt = 0; nt < 2; ++nt) {
    const int col = c0 + wn * 32 + nt * 16 + l;
    const int bb  = col >> 6;
    const int e   = col & 63;
    const float bv = bias[e];
#pragma unroll
    for (int mt = 0; mt < 4; ++mt) {
      const int mbase = m0 + wm * 64 + mt * 16 + h * 8;
#pragma unroll
      for (int r = 0; r < 8; ++r) {
        out[((size_t)bb * NN + (mbase + r)) * EEC + e] = acc[mt][nt][r] + bv;
      }
    }
  }
}

// ---------------------------------------------------------------------------
extern "C" void kernel_launch(void* const* d_in, const int* in_sizes, int n_in,
                              void* d_out, int out_size, void* d_ws, size_t ws_size,
                              hipStream_t stream) {
  const float* x    = (const float*)d_in[0];  // [1024,1024,64] f32
  const float* adj  = (const float*)d_in[1];  // [1024,1024]    f32
  const float* W    = (const float*)d_in[2];  // [64,64]        f32
  const float* avec = (const float*)d_in[3];  // [64]           f32
  const float* bias = (const float*)d_in[4];  // [64]           f32
  float* out = (float*)d_out;                 // [1024,1024,64] f32

  char* ws = (char*)d_ws;
  float*    svec = (float*)ws;                          // 4 MB
  _Float16* attw = (_Float16*)(ws + ((size_t)4 << 20)); // 2 MB
  _Float16* wx2  = (_Float16*)(ws + ((size_t)8 << 20)); // 128 MB

  k_wx<<<dim3(16, 1024), 256, 0, stream>>>(x, W, avec, wx2, svec);
  k_softmax<<<dim3(1024), 256, 0, stream>>>(svec, adj, attw);
  k_gemm<<<dim3(512, 8), 256, 0, stream>>>(attw, wx2, bias, out);
}
